// MAHABlock_15659450761671
// MI455X (gfx1250) — matrix operand order repaired and checked by reading, not compiled, not run on hardware
//
#include <hip/hip_runtime.h>
#include <hip/hip_bf16.h>
#include <math.h>

// ---------------------------------------------------------------------------
// MAHA block for MI455X (gfx1250, wave32, WMMA + TDM).
// B=4, L=1024, D=1024, H=16, F=4096, S=4, CR=2, dh=64
// - All matmuls on v_wmma_f32_16x16x32_bf16 (f32 accumulate).
// - Plain-GEMM tiles staged by the Tensor Data Mover (tensor_load_to_lds),
//   then a cooperative LDS->LDS pass converts f32->bf16 and transposes B so
//   every WMMA fragment read is a contiguous b128 DS load.
// - Conv taps staged directly (shift/mask indexing, pair-packed bf16 stores).
// - Attention is fused flash-style; K/V tiles also TDM-staged.
// ---------------------------------------------------------------------------

#define B_ 4
#define L_ 1024
#define D_ 1024
#define H_ 16
#define F_ 4096
#define S_ 4
#define CR_ 2
#define DH_ 64

typedef __attribute__((ext_vector_type(16))) __bf16 v16bf;
typedef __attribute__((ext_vector_type(8)))  float  v8f;
typedef __attribute__((ext_vector_type(4)))  unsigned int u32x4;
typedef __attribute__((ext_vector_type(8)))  int i32x8;
typedef __attribute__((ext_vector_type(4)))  int i32x4;

__device__ __forceinline__ __bf16 tobf(float f) { return (__bf16)f; }

// pack two floats into two bf16 in one dword (for b32 LDS stores)
__device__ __forceinline__ unsigned int pk2(float a, float b) {
  union { __bf16 h[2]; unsigned int u; } x;
  x.h[0] = (__bf16)a; x.h[1] = (__bf16)b;
  return x.u;
}

__device__ __forceinline__ unsigned int lds_off(const void* p) {
  return (unsigned int)(uintptr_t)p;   // flat LDS aperture: addr[31:0] = LDS offset
}

// ---------------------------------------------------------------------------
// Tensor Data Mover: 2D tile load, global f32 -> LDS. D# per ISA ch.8:
//  group0: count=1 | lds_addr | global_addr[56:0] | type=2
//  group1: data_size=4B, tensor_dim0/1 (OOB), tile_dim0/1, tensor_dim0_stride
// This toolchain exposes the 6-arg builtin:
//  (u32x4 g0, i32x8 g1, i32x4 g2, i32x4 g3, i32x8 pad, i32 cpol)
// ---------------------------------------------------------------------------
__device__ __forceinline__ void tdm_load_2d_f32(const float* gptr, unsigned int ldsa,
                                                unsigned long long dim0,
                                                unsigned long long dim1,
                                                unsigned int tile0, unsigned int tile1,
                                                unsigned long long stride0)
{
  unsigned long long ga = (unsigned long long)(uintptr_t)gptr;
  u32x4 g0;
  g0[0] = 1u;                                        // count=1, user mode
  g0[1] = ldsa;                                      // lds_addr
  g0[2] = (unsigned int)(ga & 0xFFFFFFFFu);          // global_addr[31:0]
  g0[3] = (unsigned int)((ga >> 32) & 0x1FFFFFFu)    // global_addr[56:32]
        | (2u << 30);                                // type=2 ("image")
  i32x8 g1;
  g1[0] = (int)(2u << 16);                           // data_size=2 (4 bytes)
  g1[1] = (int)((unsigned int)(dim0 << 16));         // tensor_dim0[15:0] @bit48
  g1[2] = (int)(((unsigned int)(dim0 >> 16) & 0xFFFFu)
        | (((unsigned int)dim1 & 0xFFFFu) << 16));   // dim0[31:16] | dim1[15:0]
  g1[3] = (int)((((unsigned int)(dim1 >> 16)) & 0xFFFFu)
        | (tile0 << 16));                            // dim1[31:16] | tile_dim0
  g1[4] = (int)(tile1 & 0xFFFFu);                    // tile_dim1 (tile_dim2=0)
  g1[5] = (int)(unsigned int)(stride0 & 0xFFFFFFFFu);
  g1[6] = (int)(unsigned int)((stride0 >> 32) & 0xFFFFu); // stride0[47:32], stride1=0
  g1[7] = 0;
  i32x4 z4 = {0, 0, 0, 0};
  i32x8 z8 = {0, 0, 0, 0, 0, 0, 0, 0};
  __builtin_amdgcn_tensor_load_to_lds(g0, g1, z4, z4, z8, 0);
}

// ===========================================================================
// Generic GEMM:  C[M,N] (+)= act( A[M,K] @ B[K,N] + bias )
//  convW=0 (plain): TDM-staged f32 tiles -> convert/transpose pass -> WMMA.
//  convW=1 (tapped conv): A gathered with per-tap row offsets; B gathered from
//    OIH conv weight W[o,i,t]. 32-wide K slab never crosses a tap (32|1024),
//    so tap index/offset are uniform per K-step. 128-row M tile never crosses
//    a batch (128 | Lout), so the batch index is uniform per block.
// Tile: 128x128x32, 256 threads (8 waves, 4x2 grid, 32x64 per wave).
// ===========================================================================
#define ASTR 40   // bf16 LDS row stride (16B aligned)

__global__ __launch_bounds__(256)
void gemm_conv_kernel(const float* __restrict__ A, const float* __restrict__ Bw,
                      const float* __restrict__ bias, float* __restrict__ C,
                      int M, int N, int K, int lda, int ldb, int ldc,
                      int Lin, int loutShift, int rowShift,
                      int kShift, int kMask,
                      int off0, int off1, int off2,
                      int T, int convW, int act, int accum)
{
  __shared__ __align__(16) float  Afs[128 * 32];   // TDM landing pad (f32)
  __shared__ __align__(16) float  Bfs[32 * 128];   // TDM landing pad (f32)
  __shared__ __align__(16) __bf16 As_[128 * ASTR]; // A tile, row-major  [m][k]
  __shared__ __align__(16) __bf16 BsT[128 * ASTR]; // B tile, transposed [n][k]

  const int tid  = threadIdx.x;
  const int lane = tid & 31;
  const int wave = tid >> 5;
  const int wm   = wave & 3;
  const int wn   = wave >> 2;
  const int ln   = lane & 15;
  const int lh   = lane >> 4;
  const int tileM = blockIdx.y * 128;
  const int tileN = blockIdx.x * 128;

  // conv path: batch index is uniform per tile
  const int b  = tileM >> loutShift;
  const int l0 = tileM - (b << loutShift);
  const float* Abase = A + ((size_t)b * Lin) * lda;

  v8f acc[2][4];
#pragma unroll
  for (int i = 0; i < 2; ++i)
#pragma unroll
    for (int j = 0; j < 4; ++j) acc[i][j] = (v8f){0,0,0,0,0,0,0,0};

  for (int k0 = 0; k0 < K; k0 += 32) {
    __syncthreads();
    if (!convW) {
      // ---- TDM: DMA both f32 tiles into LDS (one wave issues, all consume)
      if (wave == 0) {
        tdm_load_2d_f32(A + (size_t)tileM * lda + k0, lds_off(Afs),
                        (unsigned long long)lda, (unsigned long long)M,
                        32u, 128u, (unsigned long long)lda);
        tdm_load_2d_f32(Bw + (size_t)k0 * ldb + tileN, lds_off(Bfs),
                        (unsigned long long)ldb, (unsigned long long)K,
                        128u, 32u, (unsigned long long)ldb);
        __builtin_amdgcn_s_wait_tensorcnt(0);
      }
      __syncthreads();
      // ---- convert f32->bf16; transpose B. 2048 dword-pairs each, 8/thread.
      for (int idx = tid; idx < 2048; idx += 256) {
        int ar = idx >> 4, ac2 = (idx & 15) * 2;
        *(unsigned int*)&As_[ar * ASTR + ac2] =
            pk2(Afs[ar * 32 + ac2], Afs[ar * 32 + ac2 + 1]);
        int bc = idx >> 4, br2 = (idx & 15) * 2;
        *(unsigned int*)&BsT[bc * ASTR + br2] =
            pk2(Bfs[br2 * 128 + bc], Bfs[(br2 + 1) * 128 + bc]);
      }
    } else {
      // ---- tapped conv staging: tap uniform across this K slab
      const int t   = k0 >> kShift;
      const int iib = k0 & kMask;
      const int off = (t == 0) ? off0 : ((t == 1) ? off1 : off2);
      for (int idx = tid; idx < 2048; idx += 256) {
        int ar = idx >> 4, ac2 = (idx & 15) * 2;
        int il = ((l0 + ar) << rowShift) + off;
        float v0 = 0.f, v1 = 0.f;
        if ((unsigned)il < (unsigned)Lin) {
          const float* p = Abase + (size_t)il * lda + iib + ac2;
          v0 = p[0]; v1 = p[1];
        }
        *(unsigned int*)&As_[ar * ASTR + ac2] = pk2(v0, v1);
        int bc = idx >> 4, br2 = (idx & 15) * 2;
        const float* q = Bw + (size_t)(tileN + bc) * K + (size_t)(iib + br2) * T + t;
        *(unsigned int*)&BsT[bc * ASTR + br2] = pk2(q[0], q[T]);
      }
    }
    __syncthreads();

    // ---- fragments (all contiguous 16xbf16 runs -> b128 DS loads) + WMMA
    v16bf af[2], bfr[4];
#pragma unroll
    for (int mf = 0; mf < 2; ++mf) {
      const __bf16* p = &As_[(wm * 32 + mf * 16 + ln) * ASTR];
#pragma unroll
      for (int e = 0; e < 8; ++e) af[mf][e]     = p[lh * 16 + e];
#pragma unroll
      for (int e = 0; e < 8; ++e) af[mf][8 + e] = p[lh * 16 + 8 + e];
    }
#pragma unroll
    for (int nf = 0; nf < 4; ++nf) {
      const __bf16* p = &BsT[(wn * 64 + nf * 16 + ln) * ASTR + lh * 16];
#pragma unroll
      for (int e = 0; e < 16; ++e) bfr[nf][e] = p[e];
    }
#pragma unroll
    for (int mf = 0; mf < 2; ++mf)
#pragma unroll
      for (int nf = 0; nf < 4; ++nf)
        acc[mf][nf] = __builtin_amdgcn_wmma_f32_16x16x32_bf16(
            false, af[mf], false, bfr[nf], (short)0, acc[mf][nf], false, false);
  }

  // ---- epilogue (tiles are exact: M,N multiples of 128)
#pragma unroll
  for (int mf = 0; mf < 2; ++mf)
#pragma unroll
    for (int nf = 0; nf < 4; ++nf)
#pragma unroll
      for (int r = 0; r < 8; ++r) {
        int row = tileM + wm * 32 + mf * 16 + r + 8 * lh;
        int col = tileN + wn * 64 + nf * 16 + ln;
        float v = acc[mf][nf][r];
        if (bias) v += bias[col];
        if (act == 1) v = fmaxf(v, 0.f);
        else if (act == 2) v = 0.5f * v * (1.f + erff(v * 0.70710678118654752f));
        size_t ci = (size_t)row * ldc + col;
        if (accum) C[ci] += v; else C[ci] = v;
      }
}

// ===========================================================================
// Flash attention, one scale. Block = 128 threads (4 waves); each wave owns a
// 16-row Q tile for one (b,h). K/V 32x64 f32 tiles TDM-staged, converted to
// bf16 (V transposed) so all WMMA fragment reads are contiguous.
// ===========================================================================
#define KSTR 72   // Ks row stride (bf16)
#define VSTR 40   // VsT row stride (bf16)

__global__ __launch_bounds__(128)
void flash_attn_kernel(const float* __restrict__ Q, const float* __restrict__ Kb,
                       const float* __restrict__ Vb, float* __restrict__ Oacc,
                       const float* __restrict__ wS, int Ls, float scale)
{
  const int lane = threadIdx.x & 31;
  const int wave = threadIdx.x >> 5;
  const int h = blockIdx.y;
  const int b = blockIdx.z;
  const int qtile = blockIdx.x * 4 + wave;   // 16 Q rows per wave
  const int ln = lane & 15;
  const int lh = lane >> 4;

  __shared__ __align__(16) float  Kf[32 * 64];       // TDM landing pads
  __shared__ __align__(16) float  Vf[32 * 64];
  __shared__ __align__(16) __bf16 Ks[32 * KSTR];     // [kj][d]
  __shared__ __align__(16) __bf16 VsT[64 * VSTR];    // [d][kj]
  __shared__ __align__(16) __bf16 Ps[4][16 * 32];    // per-wave P tile

  // Q fragments: 16x32 x2 (dh = 64)
  const float* qbase = Q + ((size_t)b * L_ + qtile * 16 + ln) * D_ + h * DH_;
  v16bf qa[2];
#pragma unroll
  for (int f = 0; f < 2; ++f) {
#pragma unroll
    for (int e = 0; e < 8; ++e) qa[f][e]     = tobf(qbase[f * 32 + lh * 16 + e]);
#pragma unroll
    for (int e = 0; e < 8; ++e) qa[f][8 + e] = tobf(qbase[f * 32 + lh * 16 + 8 + e]);
  }

  v8f o[4];
#pragma unroll
  for (int n = 0; n < 4; ++n) o[n] = (v8f){0,0,0,0,0,0,0,0};
  float mrow[8], lrow[8];
#pragma unroll
  for (int r = 0; r < 8; ++r) { mrow[r] = -INFINITY; lrow[r] = 0.f; }

  for (int kb = 0; kb < Ls; kb += 32) {
    __syncthreads();
    if (wave == 0) {
      const float* kg = Kb + ((size_t)b * Ls + kb) * D_ + h * DH_;
      const float* vg = Vb + ((size_t)b * Ls + kb) * D_ + h * DH_;
      tdm_load_2d_f32(kg, lds_off(Kf), (unsigned long long)D_, 32u, 64u, 32u,
                      (unsigned long long)D_);
      tdm_load_2d_f32(vg, lds_off(Vf), (unsigned long long)D_, 32u, 64u, 32u,
                      (unsigned long long)D_);
      __builtin_amdgcn_s_wait_tensorcnt(0);
    }
    __syncthreads();
    // convert: 1024 dword-pairs for K (row-major) and V (transposed)
    for (int idx = threadIdx.x; idx < 1024; idx += 128) {
      int kr = idx >> 5, kc2 = (idx & 31) * 2;
      *(unsigned int*)&Ks[kr * KSTR + kc2] =
          pk2(Kf[kr * 64 + kc2], Kf[kr * 64 + kc2 + 1]);
      int d = idx >> 4, kj2 = (idx & 15) * 2;
      *(unsigned int*)&VsT[d * VSTR + kj2] =
          pk2(Vf[kj2 * 64 + d], Vf[(kj2 + 1) * 64 + d]);
    }
    __syncthreads();

    // S = Q @ K^T   (two 16x16 column groups, chained over dh)
    v8f sfr[2];
#pragma unroll
    for (int g = 0; g < 2; ++g) {
      v8f s = (v8f){0,0,0,0,0,0,0,0};
#pragma unroll
      for (int f = 0; f < 2; ++f) {
        v16bf bb;
        const __bf16* kp = &Ks[(g * 16 + ln) * KSTR + f * 32 + lh * 16];
#pragma unroll
        for (int e = 0; e < 16; ++e) bb[e] = kp[e];
        s = __builtin_amdgcn_wmma_f32_16x16x32_bf16(false, qa[f], false, bb,
                                                    (short)0, s, false, false);
      }
      sfr[g] = s;
    }

    // online softmax: rows live on (lh, vgpr r); columns across 16 lanes
    float fac[8];
#pragma unroll
    for (int r = 0; r < 8; ++r) {
      float s0 = sfr[0][r] * scale, s1 = sfr[1][r] * scale;
      float mt = fmaxf(s0, s1);
      for (int off = 1; off < 16; off <<= 1) mt = fmaxf(mt, __shfl_xor(mt, off, 32));
      float mn = fmaxf(mrow[r], mt);
      fac[r] = expf(mrow[r] - mn);
      float p0 = expf(s0 - mn), p1 = expf(s1 - mn);
      sfr[0][r] = p0; sfr[1][r] = p1;
      float lt = p0 + p1;
      for (int off = 1; off < 16; off <<= 1) lt += __shfl_xor(lt, off, 32);
      lrow[r] = lrow[r] * fac[r] + lt;
      mrow[r] = mn;
    }
#pragma unroll
    for (int n = 0; n < 4; ++n)
#pragma unroll
      for (int r = 0; r < 8; ++r) o[n][r] *= fac[r];

    // P (C-layout) -> per-wave LDS -> A-layout (DS is in-order per wave)
    __bf16* pp = &Ps[wave][0];
#pragma unroll
    for (int g = 0; g < 2; ++g)
#pragma unroll
      for (int r = 0; r < 8; ++r)
        pp[(r + 8 * lh) * 32 + g * 16 + ln] = tobf(sfr[g][r]);
    asm volatile("s_wait_dscnt 0x0" ::: "memory");
    v16bf pa;
    {
      const __bf16* pr = &pp[ln * 32];
#pragma unroll
      for (int e = 0; e < 8; ++e) pa[e]     = pr[lh * 16 + e];
#pragma unroll
      for (int e = 0; e < 8; ++e) pa[8 + e] = pr[lh * 16 + 8 + e];
    }

    // O += P @ V
#pragma unroll
    for (int n = 0; n < 4; ++n) {
      v16bf vb;
      const __bf16* vp = &VsT[(n * 16 + ln) * VSTR + lh * 16];
#pragma unroll
      for (int e = 0; e < 16; ++e) vb[e] = vp[e];
      o[n] = __builtin_amdgcn_wmma_f32_16x16x32_bf16(false, pa, false, vb,
                                                     (short)0, o[n], false, false);
    }
  }

  const float ws = wS[0];
#pragma unroll
  for (int n = 0; n < 4; ++n)
#pragma unroll
    for (int r = 0; r < 8; ++r) {
      int row = qtile * 16 + r + 8 * lh;
      int col = h * DH_ + n * 16 + ln;
      size_t gi = ((size_t)b * L_ + row) * D_ + col;
      Oacc[gi] += ws * (o[n][r] / lrow[r]);
    }
}

// ===========================================================================
// Elementwise / reduction helpers
// ===========================================================================
__global__ __launch_bounds__(256)
void addln_kernel(const float* __restrict__ a, const float* __restrict__ bv,
                  const float* __restrict__ extra, const float* __restrict__ g,
                  const float* __restrict__ bt, float* __restrict__ out)
{
  const int row = blockIdx.x;
  const int tid = threadIdx.x;
  __shared__ float red[256];
  const size_t base = (size_t)row * D_;
  float v[4];
  float s = 0.f;
#pragma unroll
  for (int j = 0; j < 4; ++j) {
    int c = tid + j * 256;
    float x = a[base + c] + bv[base + c];
    if (extra) x += extra[c];
    v[j] = x; s += x;
  }
  red[tid] = s; __syncthreads();
  for (int st = 128; st > 0; st >>= 1) { if (tid < st) red[tid] += red[tid + st]; __syncthreads(); }
  float mu = red[0] * (1.f / D_);
  __syncthreads();
  float sv = 0.f;
#pragma unroll
  for (int j = 0; j < 4; ++j) { float d = v[j] - mu; sv += d * d; }
  red[tid] = sv; __syncthreads();
  for (int st = 128; st > 0; st >>= 1) { if (tid < st) red[tid] += red[tid + st]; __syncthreads(); }
  float rs = rsqrtf(red[0] * (1.f / D_) + 1e-5f);
#pragma unroll
  for (int j = 0; j < 4; ++j) {
    int c = tid + j * 256;
    out[base + c] = (v[j] - mu) * rs * g[c] + bt[c];
  }
}

__global__ void zero_kernel(float* __restrict__ p, size_t n)
{
  size_t i = (size_t)blockIdx.x * blockDim.x + threadIdx.x;
  size_t stride = (size_t)gridDim.x * blockDim.x;
  for (; i < n; i += stride) p[i] = 0.f;
}

__global__ void agg_kernel(const float* __restrict__ logits,
                           float* __restrict__ w, float* __restrict__ aux)
{
  if (threadIdx.x == 0 && blockIdx.x == 0) {
    float mx = -INFINITY;
    for (int i = 0; i < S_; ++i) mx = fmaxf(mx, logits[i]);
    float e[S_], s = 0.f;
    for (int i = 0; i < S_; ++i) { e[i] = expf(logits[i] - mx); s += e[i]; }
    float al = 0.f;
    for (int i = 0; i < S_; ++i) {
      float wi = e[i] / s;
      w[i] = wi;
      al -= wi * logf(wi + 1e-9f);
    }
    *aux = al;
  }
}

// ===========================================================================
// Host-side launch
// ===========================================================================
static inline dim3 gemm_grid(int M, int N) {
  return dim3((N + 127) / 128, (M + 127) / 128);
}

extern "C" void kernel_launch(void* const* d_in, const int* in_sizes, int n_in,
                              void* d_out, int out_size, void* d_ws, size_t ws_size,
                              hipStream_t stream)
{
  const float* x      = (const float*)d_in[0];
  const float* W_dc   = (const float*)d_in[1];
  const float* b_dc   = (const float*)d_in[2];
  const float* W_dec  = (const float*)d_in[3];
  const float* Wq     = (const float*)d_in[4];
  const float* Wk     = (const float*)d_in[5];
  const float* Wv     = (const float*)d_in[6];
  const float* Wo     = (const float*)d_in[7];
  const float* aggl   = (const float*)d_in[8];
  const float* gamma1 = (const float*)d_in[9];
  const float* beta1  = (const float*)d_in[10];
  const float* gamma2 = (const float*)d_in[11];
  const float* beta2  = (const float*)d_in[12];
  const float* W1     = (const float*)d_in[13];
  const float* b1     = (const float*)d_in[14];
  const float* W2     = (const float*)d_in[15];
  const float* b2     = (const float*)d_in[16];
  float* out = (float*)d_out;

  // workspace layout (floats)
  float* ws = (float*)d_ws;
  const size_t n_xc = (size_t)B_ * L_ * D_;           // 4 Mi
  float* xc   = ws;                                    // [B,L,D]
  float* sc1  = xc  + n_xc;                            // [B,512,D]
  float* sc2  = sc1 + (size_t)B_ * 512 * D_;           // [B,256,D]
  float* sc3  = sc2 + (size_t)B_ * 256 * D_;           // [B,128,D]
  float* qb   = sc3 + (size_t)B_ * 128 * D_;           // [B,L,D]
  float* kb   = qb  + n_xc;
  float* vb   = kb  + n_xc;
  float* aacc = vb  + n_xc;                            // attention accumulator
  float* wagg = aacc + n_xc;                           // softmax(agg_logits) [4]
  float* maha = qb;                                    // reuse q
  float* x1   = kb;                                    // reuse k
  float* facc = vb;                                    // reuse v (FFN accumulator)
  float* hid  = aacc;                                  // reuse attn acc (FFN hidden)

  const dim3 blk256(256), blk128(128);
  const int PLAIN_SH = 30, PLAIN_MSK = 0x3FFFFFFF;

  // 1. aggregation weights + aux loss
  agg_kernel<<<1, 32, 0, stream>>>(aggl, wagg, out + n_xc);

  // 2. dilated conv (taps -2,0,+2, K=3D) + bias + ReLU  ->  xc
  gemm_conv_kernel<<<gemm_grid(B_*L_, D_), blk256, 0, stream>>>(
      x, W_dc, b_dc, xc, B_*L_, D_, 3*D_, D_, 0, D_,
      /*Lin*/L_, /*loutShift*/10, /*rowShift*/0, /*kShift*/10, /*kMask*/1023,
      -2, 0, 2, /*T*/3, /*convW*/1, /*act relu*/1, 0);

  // 3. hierarchical decomposition (stride-2, k=2 convs, K=2D)
  gemm_conv_kernel<<<gemm_grid(B_*512, D_), blk256, 0, stream>>>(
      xc, W_dec + 0*(size_t)D_*D_*CR_, nullptr, sc1, B_*512, D_, 2*D_, D_, 0, D_,
      1024, 9, 1, 10, 1023, 0, 1, 0, 2, 1, 0, 0);
  gemm_conv_kernel<<<gemm_grid(B_*256, D_), blk256, 0, stream>>>(
      sc1, W_dec + 1*(size_t)D_*D_*CR_, nullptr, sc2, B_*256, D_, 2*D_, D_, 0, D_,
      512, 8, 1, 10, 1023, 0, 1, 0, 2, 1, 0, 0);
  gemm_conv_kernel<<<gemm_grid(B_*128, D_), blk256, 0, stream>>>(
      sc2, W_dec + 2*(size_t)D_*D_*CR_, nullptr, sc3, B_*128, D_, 2*D_, D_, 0, D_,
      256, 7, 1, 10, 1023, 0, 1, 0, 2, 1, 0, 0);

  // 4. multiscale flash attention, accumulated with convex weights
  zero_kernel<<<2048, blk256, 0, stream>>>(aacc, n_xc);
  const float* scales[S_] = { xc, sc1, sc2, sc3 };
  const int    Lss[S_]    = { 1024, 512, 256, 128 };
  const float inv_sqrt_dh = 0.125f;  // 1/sqrt(64)
  for (int s = 0; s < S_; ++s) {
    const int Ls = Lss[s];
    gemm_conv_kernel<<<gemm_grid(B_*L_, D_), blk256, 0, stream>>>(
        xc, Wq + (size_t)s*D_*D_, nullptr, qb, B_*L_, D_, D_, D_, D_, D_,
        B_*L_, 0, 0, PLAIN_SH, PLAIN_MSK, 0, 0, 0, 1, 0, 0, 0);
    gemm_conv_kernel<<<gemm_grid(B_*Ls, D_), blk256, 0, stream>>>(
        scales[s], Wk + (size_t)s*D_*D_, nullptr, kb, B_*Ls, D_, D_, D_, D_, D_,
        B_*Ls, 0, 0, PLAIN_SH, PLAIN_MSK, 0, 0, 0, 1, 0, 0, 0);
    gemm_conv_kernel<<<gemm_grid(B_*Ls, D_), blk256, 0, stream>>>(
        scales[s], Wv, nullptr, vb, B_*Ls, D_, D_, D_, D_, D_,
        B_*Ls, 0, 0, PLAIN_SH, PLAIN_MSK, 0, 0, 0, 1, 0, 0, 0);
    flash_attn_kernel<<<dim3(L_/64, H_, B_), blk128, 0, stream>>>(
        qb, kb, vb, aacc, wagg + s, Ls, inv_sqrt_dh);
  }

  // 5. output projection: maha = aacc @ Wo (shared Wo applied post-aggregation)
  gemm_conv_kernel<<<gemm_grid(B_*L_, D_), blk256, 0, stream>>>(
      aacc, Wo, nullptr, maha, B_*L_, D_, D_, D_, D_, D_,
      B_*L_, 0, 0, PLAIN_SH, PLAIN_MSK, 0, 0, 0, 1, 0, 0, 0);

  // 6. x1 = LN(residual + maha)
  addln_kernel<<<B_*L_, blk256, 0, stream>>>(x, maha, nullptr, gamma1, beta1, x1);

  // 7. FFN, chunked over F in 512-column slices (accumulating GEMM)
  zero_kernel<<<2048, blk256, 0, stream>>>(facc, n_xc);
  for (int c = 0; c < F_ / 512; ++c) {
    gemm_conv_kernel<<<gemm_grid(B_*L_, 512), blk256, 0, stream>>>(
        x1, W1 + (size_t)c*512, b1 + (size_t)c*512, hid,
        B_*L_, 512, D_, D_, F_, 512,
        B_*L_, 0, 0, PLAIN_SH, PLAIN_MSK, 0, 0, 0, 1, 0, /*gelu*/2, 0);
    gemm_conv_kernel<<<gemm_grid(B_*L_, D_), blk256, 0, stream>>>(
        hid, W2 + (size_t)c*512*D_, nullptr, facc,
        B_*L_, D_, 512, 512, D_, D_,
        B_*L_, 0, 0, PLAIN_SH, PLAIN_MSK, 0, 0, 0, 1, 0, 0, /*accum*/1);
  }

  // 8. out = LN(x1 + ffn + b2)
  addln_kernel<<<B_*L_, blk256, 0, stream>>>(x1, facc, b2, gamma2, beta2, out);
}
